// CRFAspectSent_67473936220639
// MI455X (gfx1250) — matrix operand order repaired and checked by reading, not compiled, not run on hardware
//
#include <hip/hip_runtime.h>
#include <hip/hip_bf16.h>
#include <math.h>

typedef __attribute__((ext_vector_type(16))) _Float16 v16h;
typedef __attribute__((ext_vector_type(8)))  _Float16 v8h;
typedef __attribute__((ext_vector_type(8)))  float    v8f;

#define B_    128
#define T_    256
#define HDIR  256
#define HH    512      // 2*HDIR
#define GCOL  1024     // 4*HDIR
#define WDIM  300
#define MDIM  50
#define NKX   11       // ceil(350/32) -> K 0..351 (pad 350,351 = 0)
#define NKH   8        // 256/32
#define NKS   19       // total K-steps (608/32)
#define NT    64       // 1024/16 N tiles

// ---- workspace layout (bytes, all 256-aligned) ----
static const size_t OFF_WF     = 0;                                  // 19*64*512 halves
static const size_t OFF_WB     = OFF_WF  + (size_t)NKS*NT*512*2;
static const size_t OFF_BF     = OFF_WB  + (size_t)NKS*NT*512*2;
static const size_t OFF_BB     = OFF_BF  + 4096;
static const size_t OFF_X      = OFF_BB  + 4096;                     // 8*256*11*512 halves
static const size_t OFF_CTX    = OFF_X   + (size_t)8*T_*NKX*512*2;
static const size_t OFF_TRI    = OFF_CTX + (size_t)B_*T_*HH*4;
static const size_t OFF_ALPHA  = OFF_TRI + (size_t)B_*T_*2*4;
static const size_t OFF_SEL    = OFF_ALPHA + (size_t)B_*T_*2*4;
static const size_t OFF_SELSUM = OFF_SEL + (size_t)B_*T_*4;
static const size_t OFF_SV     = OFF_SELSUM + 512;
static const size_t OFF_SC     = OFF_SV  + (size_t)B_*HH*4;
static const size_t OFF_XWF    = OFF_SC  + 1536 + 256;               // x@Wih partials, f16
static const size_t OFF_XWB    = OFF_XWF + (size_t)B_*T_*GCOL*2;
static const size_t OFF_END    = OFF_XWB + (size_t)B_*T_*GCOL*2;

__device__ __forceinline__ float fast_tanh(float x) {
  x = fminf(fmaxf(x, -15.f), 15.f);
  float e = __expf(2.f * x);
  return (e - 1.f) / (e + 1.f);
}
__device__ __forceinline__ float fast_sig(float x) {
  return 1.f / (1.f + __expf(-x));
}

// ================= weight prep: f32 -> f16, gate-interleaved perm, =========
// ================= stored in per-lane WMMA B-fragment layout ================
// W_sw flat = ((ks*64 + ntile)*32 + lane)*16 + h
// value = Wrow[K = ks*32 + (h<8 ? (lane<16?0:8)+h : 16+(lane<16?0:8)+h-8)] at permuted column ntile*16+(lane&15)
__global__ __launch_bounds__(256) void prep_w(
    const float* __restrict__ WihF, const float* __restrict__ WhhF,
    const float* __restrict__ WihB, const float* __restrict__ WhhB,
    _Float16* __restrict__ WF, _Float16* __restrict__ WB)
{
  size_t id = (size_t)blockIdx.x * 256 + threadIdx.x;
  const size_t per = (size_t)NKS * NT * 512;
  if (id >= 2 * per) return;
  int dir = (int)(id / per);
  size_t r = id % per;
  int h    = (int)(r & 15);
  int lane = (int)((r >> 4) & 31);
  size_t q = r >> 9;
  int ntile = (int)(q % NT);
  int ks    = (int)(q / NT);
  int K = ks * 32 + ((h < 8) ? (((lane < 16) ? 0 : 8) + h) : (16 + ((lane < 16) ? 0 : 8) + (h - 8)));
  int p = ntile * 16 + (lane & 15);          // permuted column
  int grp = p >> 6, rem = p & 63, gate = rem >> 4, nn = rem & 15;
  int orig = gate * HDIR + grp * 16 + nn;    // original gate-major column
  const float* Wih = dir ? WihB : WihF;
  const float* Whh = dir ? WhhB : WhhF;
  float v = 0.f;
  if (K < WDIM + MDIM)      v = Wih[(size_t)K * GCOL + orig];
  else if (K >= 352)        v = Whh[(size_t)(K - 352) * GCOL + orig];
  (dir ? WB : WF)[r] = (_Float16)v;
}

__global__ __launch_bounds__(256) void prep_bias(
    const float* __restrict__ bf, const float* __restrict__ bb,
    float* __restrict__ BF, float* __restrict__ BB)
{
  int id = blockIdx.x * 256 + threadIdx.x;
  if (id >= 2048) return;
  int dir = id >> 10, p = id & 1023;
  int grp = p >> 6, rem = p & 63, gate = rem >> 4, nn = rem & 15;
  int orig = gate * HDIR + grp * 16 + nn;
  (dir ? BB : BF)[p] = (dir ? bb : bf)[orig];
}

// ========= embedding gather, f16, pre-swizzled A-fragment layout ===========
// X_sw flat = (((g*T + t)*NKX + ks)*32 + lane)*16 + h ; lane row m = lane&15
__global__ __launch_bounds__(256) void prep_x(
    const int* __restrict__ sents, const int* __restrict__ masks,
    const float* __restrict__ wemb, const float* __restrict__ memb,
    _Float16* __restrict__ Xsw)
{
  size_t id = (size_t)blockIdx.x * 256 + threadIdx.x;
  const size_t tot = (size_t)8 * T_ * NKX * 512;
  if (id >= tot) return;
  int h    = (int)(id & 15);
  int lane = (int)((id >> 4) & 31);
  size_t r = id >> 9;
  int ks = (int)(r % NKX); r /= NKX;
  int t  = (int)(r % T_);
  int g  = (int)(r / T_);
  int m  = lane & 15;
  int koff = (lane < 16) ? 0 : 8;
  int kk = (h < 8) ? (koff + h) : (16 + koff + (h - 8));
  int K = ks * 32 + kk;
  int b = g * 16 + m;
  float v = 0.f;
  if (K < WDIM)             v = wemb[(size_t)sents[b * T_ + t] * WDIM + K];
  else if (K < WDIM + MDIM) v = memb[(size_t)masks[b * T_ + t] * MDIM + (K - WDIM)];
  Xsw[id] = (_Float16)v;
}

// ============ full-GPU WMMA GEMM: xw[g,t] = X[g,t] @ Wih (t-parallel) ======
// grid.x = 2*8*T: dir = blockIdx.x >> 11, gt = blockIdx.x & 2047
// software-pipelined: K-step ks+1 loads issued before consuming ks fragments
__global__ __launch_bounds__(512) void xw_gemm(
    const _Float16* __restrict__ Xsw,
    const _Float16* __restrict__ WFp, const _Float16* __restrict__ WBp,
    _Float16* __restrict__ xwF, _Float16* __restrict__ xwB)
{
  const int dir = blockIdx.x >> 11;
  const int gt  = blockIdx.x & 2047;         // g*T + t
  const _Float16* W = dir ? WBp : WFp;
  _Float16* xw      = dir ? xwB : xwF;
  const int w = threadIdx.x >> 5, lane = threadIdx.x & 31;
  const v8f vz = {0.f,0.f,0.f,0.f,0.f,0.f,0.f,0.f};
  v8f acc[4] = {vz, vz, vz, vz};
  const _Float16* xa  = Xsw + (size_t)gt * NKX * 512 + lane * 16;
  const _Float16* wb0 = W + (size_t)w * 4 * 512 + lane * 16;

  v16h a[2], bfr[2][4];
  a[0] = *(const v16h*)xa;
  #pragma unroll
  for (int t4 = 0; t4 < 4; ++t4) bfr[0][t4] = *(const v16h*)(wb0 + (size_t)t4 * 512);

  #pragma unroll
  for (int ks = 0; ks < NKX; ++ks) {
    const int pp = ks & 1;
    if (ks + 1 < NKX) {
      a[pp ^ 1] = *(const v16h*)(xa + (size_t)(ks + 1) * 512);
      const _Float16* wn = wb0 + (size_t)(ks + 1) * NT * 512;
      #pragma unroll
      for (int t4 = 0; t4 < 4; ++t4)
        bfr[pp ^ 1][t4] = *(const v16h*)(wn + (size_t)t4 * 512);
    }
    #pragma unroll
    for (int t4 = 0; t4 < 4; ++t4)
      acc[t4] = __builtin_amdgcn_wmma_f32_16x16x32_f16(
          false, a[pp], false, bfr[pp][t4], (short)0, acc[t4], false, false);
  }

  _Float16* o = xw + ((size_t)gt * 16 + w) * 4 * 256;
  #pragma unroll
  for (int t4 = 0; t4 < 4; ++t4) {
    v8h hv;
    #pragma unroll
    for (int r = 0; r < 8; ++r) hv[r] = (_Float16)acc[t4][r];
    *(v8h*)(o + t4 * 256 + lane * 8) = hv;
  }
}

// ======================= LSTM scan, WMMA recurrence ========================
// grid.x = 16: dir = blockIdx.x >> 3, batch group g = blockIdx.x & 7.
// 512 thr = 16 waves; wave w owns permuted cols [64w,64w+64) = gates i,f,g,o
// of hidden units j = 16w + (lane&15). B-fragments software-pipelined.
__global__ __launch_bounds__(512) void lstm_scan(
    const _Float16* __restrict__ Xsw,
    const _Float16* __restrict__ WFp, const _Float16* __restrict__ WBp,
    const float* __restrict__ biasF, const float* __restrict__ biasB,
    const int* __restrict__ lens, float* __restrict__ ctx,
    const _Float16* __restrict__ xwF, const _Float16* __restrict__ xwB,
    int useXW)
{
  __shared__ __align__(32) _Float16 hbuf[2][NKH][32][16];  // 16 KB, A-frag layout
  const int dir  = blockIdx.x >> 3;
  const int g    = blockIdx.x & 7;           // batch rows 16g..16g+15
  const _Float16* W    = dir ? WBp  : WFp;
  const float*    bias = dir ? biasB : biasF;
  const _Float16* xw   = dir ? xwB  : xwF;

  const int tid  = threadIdx.x;
  const int w    = tid >> 5;
  const int lane = tid & 31;
  const int n    = lane & 15;
  const int hi   = (lane >= 16) ? 1 : 0;

  for (int i = tid; i < 2 * NKH * 32 * 16; i += 512)
    ((_Float16*)hbuf)[i] = (_Float16)0.f;

  float hreg[8], creg[8];
  #pragma unroll
  for (int r = 0; r < 8; ++r) { hreg[r] = 0.f; creg[r] = 0.f; }

  float bs[4];
  #pragma unroll
  for (int t4 = 0; t4 < 4; ++t4) bs[t4] = bias[w * 64 + t4 * 16 + n];

  int lenv[8];
  #pragma unroll
  for (int r = 0; r < 8; ++r) lenv[r] = lens[g * 16 + r + 8 * hi];

  __syncthreads();

  const int colOff = dir * HDIR;
  const int j   = 16 * w + n;                // hidden unit owned by this lane
  const int ksW = w >> 1;                    // h-frag K-step this wave writes
  const int lp  = (n & 8) ? 16 : 0;          // dest lane offset in h-frag
  const int hp  = ((w & 1) << 3) | (n & 7);  // dest half index in h-frag
  const v8f vz = {0.f,0.f,0.f,0.f,0.f,0.f,0.f,0.f};
  int cur = 0;

  for (int ts = 0; ts < T_; ++ts) {
    const int t = dir ? (T_ - 1 - ts) : ts;
    // opaque copy: stop LICM from hoisting B-fragments out of the t-loop
    const _Float16* Wp = W;
    asm volatile("" : "+s"(Wp));
    const _Float16* wb0 = Wp + (size_t)w * 4 * 512 + lane * 16;

    v8f acc[4];
    if (useXW) {
      // init accumulators from precomputed input projection (f16, stream-once)
      const _Float16* xr = xw + ((size_t)(g * T_ + t) * 16 + w) * 4 * 256;
      #pragma unroll
      for (int t4 = 0; t4 < 4; ++t4) {
        v8h hv = __builtin_nontemporal_load((const v8h*)(xr + t4 * 256 + lane * 8));
        #pragma unroll
        for (int r = 0; r < 8; ++r) acc[t4][r] = (float)hv[r];
      }
    } else {
      #pragma unroll
      for (int t4 = 0; t4 < 4; ++t4) acc[t4] = vz;
      const _Float16* xa = Xsw + ((size_t)(g * T_ + t) * NKX) * 512 + lane * 16;
      v16h a[2], bfr[2][4];
      a[0] = *(const v16h*)xa;
      #pragma unroll
      for (int t4 = 0; t4 < 4; ++t4) bfr[0][t4] = *(const v16h*)(wb0 + (size_t)t4 * 512);
      #pragma unroll
      for (int ks = 0; ks < NKX; ++ks) {
        const int pp = ks & 1;
        if (ks + 1 < NKX) {
          a[pp ^ 1] = *(const v16h*)(xa + (size_t)(ks + 1) * 512);
          const _Float16* wn = wb0 + (size_t)(ks + 1) * NT * 512;
          #pragma unroll
          for (int t4 = 0; t4 < 4; ++t4)
            bfr[pp ^ 1][t4] = *(const v16h*)(wn + (size_t)t4 * 512);
        }
        #pragma unroll
        for (int t4 = 0; t4 < 4; ++t4)
          acc[t4] = __builtin_amdgcn_wmma_f32_16x16x32_f16(
              false, a[pp], false, bfr[pp][t4], (short)0, acc[t4], false, false);
      }
    }

    // recurrent part: K = 352..607, A from LDS h; B software-pipelined
    {
      const _Float16* hb0 = wb0 + (size_t)NKX * NT * 512;
      v16h bfr[2][4];
      #pragma unroll
      for (int t4 = 0; t4 < 4; ++t4) bfr[0][t4] = *(const v16h*)(hb0 + (size_t)t4 * 512);
      #pragma unroll
      for (int ks = 0; ks < NKH; ++ks) {
        const int pp = ks & 1;
        if (ks + 1 < NKH) {
          const _Float16* wn = hb0 + (size_t)(ks + 1) * NT * 512;
          #pragma unroll
          for (int t4 = 0; t4 < 4; ++t4)
            bfr[pp ^ 1][t4] = *(const v16h*)(wn + (size_t)t4 * 512);
        }
        v16h a = *(const v16h*)&hbuf[cur][ks][lane][0];
        #pragma unroll
        for (int t4 = 0; t4 < 4; ++t4)
          acc[t4] = __builtin_amdgcn_wmma_f32_16x16x32_f16(
              false, a, false, bfr[pp][t4], (short)0, acc[t4], false, false);
      }
    }

    // gates: tiles 0..3 = i,f,g,o for hidden unit j, rows m = r + 8*hi
    const int nxt = cur ^ 1;
    #pragma unroll
    for (int r = 0; r < 8; ++r) {
      const int m = r + 8 * hi;
      float iv = fast_sig(acc[0][r] + bs[0]);
      float fv = fast_sig(acc[1][r] + bs[1]);
      float gv = fast_tanh(acc[2][r] + bs[2]);
      float ov = fast_sig(acc[3][r] + bs[3]);
      float cn = fv * creg[r] + iv * gv;
      float hn = ov * fast_tanh(cn);
      bool msk = (t < lenv[r]);
      float hc = msk ? hn : hreg[r];
      float cc = msk ? cn : creg[r];
      hreg[r] = hc; creg[r] = cc;
      hbuf[nxt][ksW][m + lp][hp] = (_Float16)hc;
      const size_t bt = (size_t)(g * 16 + m) * T_ + t;
      ctx[bt * HH + colOff + j] = msk ? hn : 0.f;
    }
    __syncthreads();
    cur = nxt;
  }
}

// =================== avg pool over inv-mask + residual add =================
__global__ __launch_bounds__(512) void avg_add(
    const int* __restrict__ masks, float* __restrict__ ctx)
{
  int b = blockIdx.x, d = threadIdx.x;
  float s = 0.f, inv = 0.f;
  for (int t = 0; t < T_; ++t) {
    float iv = 1.f - (float)masks[b * T_ + t];
    inv += iv;
    s += iv * ctx[((size_t)b * T_ + t) * HH + d];
  }
  float a = s / inv;
  for (int t = 0; t < T_; ++t) ctx[((size_t)b * T_ + t) * HH + d] += a;
}

// ========================= tri = ctx @ W_tri + b_tri =======================
__global__ __launch_bounds__(256) void tri_proj(
    const float* __restrict__ ctx, const float* __restrict__ Wt,
    const float* __restrict__ bt, float* __restrict__ tri)
{
  int w = threadIdx.x >> 5, lane = threadIdx.x & 31;
  int btid = blockIdx.x * 8 + w;            // 0..B*T-1
  const float* c = ctx + (size_t)btid * HH;
  float s0 = 0.f, s1 = 0.f;
  for (int d = lane; d < HH; d += 32) {
    float cv = c[d];
    s0 += cv * Wt[d * 2 + 0];
    s1 += cv * Wt[d * 2 + 1];
  }
  #pragma unroll
  for (int off = 16; off; off >>= 1) {
    s0 += __shfl_xor(s0, off, 32);
    s1 += __shfl_xor(s1, off, 32);
  }
  if (lane == 0) {
    tri[(size_t)btid * 2 + 0] = s0 + bt[0];
    tri[(size_t)btid * 2 + 1] = s1 + bt[1];
  }
}

// ============================ 2-state CRF marginals ========================
__device__ __forceinline__ float lse2(float a, float b) {
  float m = fmaxf(a, b);
  return m + logf(__expf(a - m) + __expf(b - m));
}

__global__ __launch_bounds__(128) void crf_k(
    const float* __restrict__ tri, const int* __restrict__ lens,
    const float* __restrict__ trans, float* __restrict__ alphas,
    float* __restrict__ sel, float* __restrict__ selsum)
{
  int b = threadIdx.x;
  const float t00 = trans[0], t01 = trans[1], t10 = trans[2], t11 = trans[3];
  const float* u = tri + (size_t)b * T_ * 2;
  float* al = alphas + (size_t)b * T_ * 2;
  int L = lens[b];
  float a0 = u[0], a1 = u[1];
  al[0] = a0; al[1] = a1;
  for (int t = 1; t < T_; ++t) {
    if (t < L) {
      float n0 = u[t * 2 + 0] + lse2(a0 + t00, a1 + t10);
      float n1 = u[t * 2 + 1] + lse2(a0 + t01, a1 + t11);
      a0 = n0; a1 = n1;
    }
    al[t * 2] = a0; al[t * 2 + 1] = a1;
  }
  float logZ = lse2(a0, a1);
  float b0 = 0.f, b1 = 0.f, ss = 0.f;
  float* sl = sel + (size_t)b * T_;
  for (int t = T_ - 1; t >= 0; --t) {
    if (t < T_ - 1 && (t + 1) < L) {
      float c0 = lse2(t00 + u[(t + 1) * 2 + 0] + b0, t01 + u[(t + 1) * 2 + 1] + b1);
      float c1 = lse2(t10 + u[(t + 1) * 2 + 0] + b0, t11 + u[(t + 1) * 2 + 1] + b1);
      b0 = c0; b1 = c1;
    }
    float mval = (t < L) ? 1.f : 0.f;
    float s = __expf(al[t * 2 + 1] + b1 - logZ) * mval;
    sl[t] = s; ss += s;
  }
  selsum[b] = ss;
}

// =========================== sent_v = sel . ctx ============================
__global__ __launch_bounds__(512) void sentv_k(
    const float* __restrict__ ctx, const float* __restrict__ sel,
    float* __restrict__ sv)
{
  int b = blockIdx.x, d = threadIdx.x;
  float s = 0.f;
  for (int t = 0; t < T_; ++t)
    s += sel[b * T_ + t] * ctx[((size_t)b * T_ + t) * HH + d];
  sv[(size_t)b * HH + d] = s;
}

// ======================= scores = sent_v @ W_lab + b =======================
__global__ __launch_bounds__(32) void score_k(
    const float* __restrict__ sv, const float* __restrict__ Wl,
    const float* __restrict__ bl, float* __restrict__ sc)
{
  int b = blockIdx.x, lane = threadIdx.x;
  float s0 = 0.f, s1 = 0.f, s2 = 0.f;
  for (int d = lane; d < HH; d += 32) {
    float v = sv[(size_t)b * HH + d];
    s0 += v * Wl[d * 3 + 0]; s1 += v * Wl[d * 3 + 1]; s2 += v * Wl[d * 3 + 2];
  }
  #pragma unroll
  for (int off = 16; off; off >>= 1) {
    s0 += __shfl_xor(s0, off, 32);
    s1 += __shfl_xor(s1, off, 32);
    s2 += __shfl_xor(s2, off, 32);
  }
  if (lane == 0) {
    sc[b * 3 + 0] = s0 + bl[0];
    sc[b * 3 + 1] = s1 + bl[1];
    sc[b * 3 + 2] = s2 + bl[2];
  }
}

// ============================== final loss =================================
__global__ __launch_bounds__(128) void final_k(
    const float* __restrict__ sc, const int* __restrict__ labels,
    const float* __restrict__ selsum, const float* __restrict__ trans,
    float* __restrict__ out)
{
  __shared__ float rn[128], rs[128];
  int b = threadIdx.x;
  float s0 = sc[b * 3], s1 = sc[b * 3 + 1], s2 = sc[b * 3 + 2];
  float m = fmaxf(s0, fmaxf(s1, s2));
  float l = m + logf(__expf(s0 - m) + __expf(s1 - m) + __expf(s2 - m));
  int lab = labels[b];
  float slv = (lab == 0) ? s0 : ((lab == 1) ? s1 : s2);
  rn[b] = -(slv - l);
  rs[b] = selsum[b];
  __syncthreads();
  for (int off = 64; off; off >>= 1) {
    if (b < off) { rn[b] += rn[b + off]; rs[b] += rs[b + off]; }
    __syncthreads();
  }
  if (b == 0) {
    float cls = rn[0] / (float)B_ + 0.01f;
    float pena = fmaxf(trans[2] - trans[0], 0.f) + fmaxf(trans[1] - trans[3], 0.f);
    out[0] = cls + 0.01f * pena + 0.01f * (rs[0] / (float)B_);
  }
}

// ================================ launcher =================================
extern "C" void kernel_launch(void* const* d_in, const int* in_sizes, int n_in,
                              void* d_out, int out_size, void* d_ws, size_t ws_size,
                              hipStream_t stream) {
  const int*   sents = (const int*)d_in[0];
  const int*   masks = (const int*)d_in[1];
  const int*   labels= (const int*)d_in[2];
  const int*   lens  = (const int*)d_in[3];
  const float* wemb  = (const float*)d_in[4];
  const float* memb  = (const float*)d_in[5];
  const float* WihF  = (const float*)d_in[6];
  const float* WhhF  = (const float*)d_in[7];
  const float* bf    = (const float*)d_in[8];
  const float* WihB  = (const float*)d_in[9];
  const float* WhhB  = (const float*)d_in[10];
  const float* bb    = (const float*)d_in[11];
  const float* Wt    = (const float*)d_in[12];
  const float* bt    = (const float*)d_in[13];
  const float* Wl    = (const float*)d_in[14];
  const float* bl    = (const float*)d_in[15];
  const float* trans = (const float*)d_in[16];

  char* ws = (char*)d_ws;
  _Float16* WF  = (_Float16*)(ws + OFF_WF);
  _Float16* WB  = (_Float16*)(ws + OFF_WB);
  float*    BF  = (float*)(ws + OFF_BF);
  float*    BB  = (float*)(ws + OFF_BB);
  _Float16* Xsw = (_Float16*)(ws + OFF_X);
  float*    ctx = (float*)(ws + OFF_CTX);
  float*    tri = (float*)(ws + OFF_TRI);
  float*    alp = (float*)(ws + OFF_ALPHA);
  float*    sel = (float*)(ws + OFF_SEL);
  float*    ssm = (float*)(ws + OFF_SELSUM);
  float*    sv  = (float*)(ws + OFF_SV);
  float*    sc  = (float*)(ws + OFF_SC);
  _Float16* XWF = (_Float16*)(ws + OFF_XWF);
  _Float16* XWB = (_Float16*)(ws + OFF_XWB);

  // deterministic path choice: ws_size is fixed for a given run
  const int useXW = (ws_size >= OFF_END) ? 1 : 0;

  const size_t wTot = (size_t)2 * NKS * NT * 512;
  prep_w<<<(unsigned)((wTot + 255) / 256), 256, 0, stream>>>(WihF, WhhF, WihB, WhhB, WF, WB);
  prep_bias<<<8, 256, 0, stream>>>(bf, bb, BF, BB);
  const size_t xTot = (size_t)8 * T_ * NKX * 512;
  prep_x<<<(unsigned)((xTot + 255) / 256), 256, 0, stream>>>(sents, masks, wemb, memb, Xsw);

  if (useXW) {
    // both directions in one grid: 4096 blocks, full-GPU parallel
    xw_gemm<<<2 * 8 * T_, 512, 0, stream>>>(Xsw, WF, WB, XWF, XWB);
  }

  // both directions concurrently: 16 blocks (8 batch groups x 2 directions)
  lstm_scan<<<16, 512, 0, stream>>>(Xsw, WF, WB, BF, BB, lens, ctx, XWF, XWB, useXW);

  avg_add<<<B_, 512, 0, stream>>>(masks, ctx);
  tri_proj<<<(B_ * T_) / 8, 256, 0, stream>>>(ctx, Wt, bt, tri);
  crf_k<<<1, 128, 0, stream>>>(tri, lens, trans, alp, sel, ssm);
  sentv_k<<<B_, 512, 0, stream>>>(ctx, sel, sv);
  score_k<<<B_, 32, 0, stream>>>(sv, Wl, bl, sc);
  final_k<<<1, 128, 0, stream>>>(sc, labels, ssm, trans, (float*)d_out);
}
